// CapsuleLayer_58437325029390
// MI455X (gfx1250) — matrix operand order repaired and checked by reading, not compiled, not run on hardware
//
#include <hip/hip_runtime.h>
#include <hip/hip_bf16.h>

typedef _Float16 f16;
typedef __attribute__((ext_vector_type(16))) _Float16 v16h;
typedef __attribute__((ext_vector_type(8)))  float    v8f;

#define NN       8
#define CI       8
#define DI       8
#define CO       16
#define DOO      16
#define HH       48
#define WW       48
#define SPATIAL  (HH * WW)        // 2304
#define KW       5
#define KRAW     (DI * KW * KW)   // 200
#define KPAD     224              // 7 * 32
#define KSTEPS   7
#define M_TOTAL  256              // CO * DOO
#define BM       128
#define BN       64
#define XSTRIDE  40               // 80-byte LDS row stride: 16B-aligned, conflict-free

// ---------------------------------------------------------------------------
// Pack weights: w[ci][m=co*16+do][di][ky][kx] f32  ->  f16 Wp[ci][m][kpad]
// ---------------------------------------------------------------------------
__global__ void pack_w_kernel(const float* __restrict__ w, f16* __restrict__ wp) {
    int idx = blockIdx.x * blockDim.x + threadIdx.x;
    if (idx >= CI * M_TOTAL * KPAD) return;
    int kp = idx % KPAD;
    int m  = (idx / KPAD) % M_TOTAL;
    int ci = idx / (KPAD * M_TOTAL);
    float val = 0.f;
    if (kp < KRAW) {
        int di  = kp / 25;
        int rem = kp % 25;
        val = w[((size_t)(ci * M_TOTAL + m) * DI + di) * 25 + rem];
    }
    wp[idx] = (f16)val;
}

// ---------------------------------------------------------------------------
// Implicit-GEMM grouped conv with WMMA:
//   per group g=n*CI+ci : uhat[g][M][s] = Wp[ci][M][k] * im2col(u)[k][s]
// Block = 256 thr (8 waves). Tile: BM=128 x BN=64. Wave -> 16-row M strip x 64 N.
// A fragments come straight from global (L2-resident weights, 2x b128/lane);
// only the im2col tile is staged in (double-buffered) LDS.
// u_hat is stored f16 to halve the dominant HBM traffic.
// ---------------------------------------------------------------------------
__global__ __launch_bounds__(256)
void conv_wmma_kernel(const float* __restrict__ u, const f16* __restrict__ wp,
                      f16* __restrict__ uhat) {
    const int g     = blockIdx.z;        // n*CI + ci
    const int ci    = g % CI;
    const int mb    = blockIdx.y * BM;
    const int nb    = blockIdx.x * BN;   // spatial base (flattened h*W+w)
    const int tid   = threadIdx.x;
    const int wave  = tid >> 5;
    const int lane  = tid & 31;
    const int half  = lane >> 4;
    const int lan16 = lane & 15;

    __shared__ __align__(16) f16 XsT[2][BN][XSTRIDE];  // 10 KB (col-major im2col)

    const float* ubase = u + (size_t)g * DI * SPATIAL;

    // weight row for this lane's A fragment (row-major, K contiguous)
    const f16* warow = wp + (size_t)(ci * M_TOTAL + mb + wave * 16 + lan16) * KPAD;

    // im2col column owned by this thread for staging
    const int stg_c  = tid >> 2;           // 0..63
    const int stg_k0 = (tid & 3) * 8;      // 0,8,16,24
    const int stg_s  = nb + stg_c;
    const int stg_h  = stg_s / WW;
    const int stg_x  = stg_s % WW;

    auto stageX = [&](int ks, int buf) {
        alignas(16) f16 vals[8];
#pragma unroll
        for (int i = 0; i < 8; ++i) {
            int kk = ks * 32 + stg_k0 + i;
            float val = 0.f;
            if (kk < KRAW) {
                int di  = kk / 25;
                int rem = kk % 25;
                int hy  = stg_h + rem / 5 - 2;
                int wx  = stg_x + rem % 5 - 2;
                if (hy >= 0 && hy < HH && wx >= 0 && wx < WW)
                    val = ubase[(di * HH + hy) * WW + wx];
            }
            vals[i] = (f16)val;
        }
        *(int4*)&XsT[buf][stg_c][stg_k0] = *(const int4*)vals;
    };

    v8f acc[4] = {};                     // 4 N-tiles of 16x16 f32

    stageX(0, 0);
#pragma unroll
    for (int ks = 0; ks < KSTEPS; ++ks) {
        __syncthreads();                 // tile[ks] visible; buf (ks+1)&1 free
        if (ks + 1 < KSTEPS) {
            stageX(ks + 1, (ks + 1) & 1);        // overlap next-step gather
            __builtin_prefetch(warow + (ks + 1) * 32, 0, 0);  // global_prefetch_b8
        }
        const int buf = ks & 1;

        // ---- A fragment direct from global: elems 0-7 = K[8h..], 8-15 = K[16+8h..]
        union { v16h v; int4 q[2]; } a;
        a.q[0] = *(const int4*)&warow[ks * 32 + 8 * half];
        a.q[1] = *(const int4*)&warow[ks * 32 + 16 + 8 * half];

        // ---- B fragments: elems 0-7 = K[8h..], 8-15 = K[16+8h..] at column (2x b128, col-major LDS)
#pragma unroll
        for (int t = 0; t < 4; ++t) {
            union { v16h v; int4 q[2]; } b;
            const f16* brow = XsT[buf][t * 16 + lan16];
            b.q[0] = *(const int4*)&brow[8 * half];
            b.q[1] = *(const int4*)&brow[16 + 8 * half];
            acc[t] = __builtin_amdgcn_wmma_f32_16x16x32_f16(
                false, a.v, false, b.v, (short)0, acc[t], false, false);
        }
    }

    // ---- write C (f16): lane n = lan16, VGPR j -> M = j + 8*half
    f16* orow = uhat + (size_t)g * M_TOTAL * SPATIAL;
#pragma unroll
    for (int t = 0; t < 4; ++t) {
        int col = nb + t * 16 + lan16;
#pragma unroll
        for (int j = 0; j < 8; ++j) {
            int mrow = mb + wave * 16 + 8 * half + j;
            orow[(size_t)mrow * SPATIAL + col] = (f16)acc[t][j];
        }
    }
}

// ---------------------------------------------------------------------------
// Routing
// ---------------------------------------------------------------------------
__global__ void zero_kernel(float* __restrict__ p, int n) {
    int i = blockIdx.x * blockDim.x + threadIdx.x;
    if (i < n) p[i] = 0.f;
}

// per (n,ci,h,w): bmax = max over Co x 5x5 window of b ; cs = sum_co exp(b - bmax)
__global__ void routing_stats_kernel(const float* __restrict__ b,
                                     float* __restrict__ bmax,
                                     float* __restrict__ cs) {
    int idx = blockIdx.x * blockDim.x + threadIdx.x;   // NN*CI*SPATIAL
    if (idx >= NN * CI * SPATIAL) return;
    int s  = idx % SPATIAL;
    int nc = idx / SPATIAL;
    int h = s / WW, x = s % WW;
    const float* bb = b + (size_t)nc * CO * SPATIAL;
    float m = -__builtin_inff();
    for (int dy = -2; dy <= 2; ++dy) {
        int hy = h + dy; if (hy < 0 || hy >= HH) continue;
        for (int dx = -2; dx <= 2; ++dx) {
            int wx = x + dx; if (wx < 0 || wx >= WW) continue;
            int sp = hy * WW + wx;
            for (int co = 0; co < CO; ++co)
                m = fmaxf(m, bb[(size_t)co * SPATIAL + sp]);
        }
    }
    float sum = 0.f;
    for (int co = 0; co < CO; ++co)
        sum += __expf(bb[(size_t)co * SPATIAL + s] - m);
    bmax[idx] = m;
    cs[idx]   = sum;
}

// per (n,ci,h,w): sumc = 5x5 zero-padded box filter of cs
__global__ void routing_sumc_kernel(const float* __restrict__ cs,
                                    float* __restrict__ sumc) {
    int idx = blockIdx.x * blockDim.x + threadIdx.x;
    if (idx >= NN * CI * SPATIAL) return;
    int s  = idx % SPATIAL;
    int nc = idx / SPATIAL;
    int h = s / WW, x = s % WW;
    const float* c = cs + (size_t)nc * SPATIAL;
    float sum = 0.f;
    for (int dy = -2; dy <= 2; ++dy) {
        int hy = h + dy; if (hy < 0 || hy >= HH) continue;
        for (int dx = -2; dx <= 2; ++dx) {
            int wx = x + dx; if (wx < 0 || wx >= WW) continue;
            sum += c[hy * WW + wx];
        }
    }
    sumc[idx] = sum;
}

// per (n,co,h,w): p = sum_ci r*u_hat ; v = squash(p) ; optional b += <u_hat, v>
__global__ __launch_bounds__(256)
void routing_pv_kernel(const f16* __restrict__ uhat,
                       float* __restrict__ b,
                       const float* __restrict__ bmax,
                       const float* __restrict__ sumc,
                       float* __restrict__ vout,
                       int update_b) {
    int idx = blockIdx.x * blockDim.x + threadIdx.x;   // NN*CO*SPATIAL
    if (idx >= NN * CO * SPATIAL) return;
    int s  = idx % SPATIAL;
    int co = (idx / SPATIAL) % CO;
    int n  = idx / (SPATIAL * CO);

    float p[DOO];
#pragma unroll
    for (int d = 0; d < DOO; ++d) p[d] = 0.f;

    for (int ci = 0; ci < CI; ++ci) {
        int nc = n * CI + ci;
        float bv = b[((size_t)nc * CO + co) * SPATIAL + s];
        float r  = __expf(bv - bmax[(size_t)nc * SPATIAL + s])
                   / sumc[(size_t)nc * SPATIAL + s];
        const f16* uh = uhat + ((size_t)nc * M_TOTAL + co * DOO) * SPATIAL + s;
#pragma unroll
        for (int d = 0; d < DOO; ++d)
            p[d] += r * (float)uh[(size_t)d * SPATIAL];
    }

    float nsq = 0.f;
#pragma unroll
    for (int d = 0; d < DOO; ++d) nsq += p[d] * p[d];
    float scale = nsq / (1.f + nsq) * rsqrtf(nsq + 1e-9f);

    float v[DOO];
#pragma unroll
    for (int d = 0; d < DOO; ++d) {
        v[d] = scale * p[d];
        vout[(((size_t)n * CO + co) * DOO + d) * SPATIAL + s] = v[d];
    }

    if (update_b) {
        for (int ci = 0; ci < CI; ++ci) {
            int nc = n * CI + ci;
            const f16* uh = uhat + ((size_t)nc * M_TOTAL + co * DOO) * SPATIAL + s;
            float dot = 0.f;
#pragma unroll
            for (int d = 0; d < DOO; ++d)
                dot += (float)uh[(size_t)d * SPATIAL] * v[d];
            b[((size_t)nc * CO + co) * SPATIAL + s] += dot;
        }
    }
}

// ---------------------------------------------------------------------------
// Launch
// ---------------------------------------------------------------------------
extern "C" void kernel_launch(void* const* d_in, const int* in_sizes, int n_in,
                              void* d_out, int out_size, void* d_ws, size_t ws_size,
                              hipStream_t stream) {
    const float* u = (const float*)d_in[0];
    const float* w = (const float*)d_in[1];

    char* ws = (char*)d_ws;
    // workspace layout (bytes)
    const size_t off_wp   = 0;                                   // 917,504
    const size_t off_uhat = 1u << 20;                            // 75,497,472 (f16)
    const size_t off_b    = off_uhat + (size_t)NN * CI * M_TOTAL * SPATIAL * 2;
    const size_t off_bmax = off_b    + (size_t)NN * CI * CO * SPATIAL * 4;
    const size_t off_cs   = off_bmax + (size_t)NN * CI * SPATIAL * 4;
    const size_t off_sumc = off_cs   + (size_t)NN * CI * SPATIAL * 4;

    f16*   wp   = (f16*)  (ws + off_wp);
    f16*   uhat = (f16*)  (ws + off_uhat);
    float* b    = (float*)(ws + off_b);
    float* bmax = (float*)(ws + off_bmax);
    float* cs   = (float*)(ws + off_cs);
    float* sumc = (float*)(ws + off_sumc);
    float* vout = (float*)d_out;

    // 1) pack weights to f16 (K padded to 224)
    {
        int tot = CI * M_TOTAL * KPAD;
        pack_w_kernel<<<(tot + 255) / 256, 256, 0, stream>>>(w, wp);
    }
    // 2) grouped conv as WMMA implicit GEMM -> uhat[g][M][s] (f16)
    {
        dim3 grid(SPATIAL / BN, M_TOTAL / BM, NN * CI);   // 36 x 2 x 64
        conv_wmma_kernel<<<grid, 256, 0, stream>>>(u, wp, uhat);
    }
    // 3) b = 0
    {
        int tot = NN * CI * CO * SPATIAL;
        zero_kernel<<<(tot + 255) / 256, 256, 0, stream>>>(b, tot);
    }
    // 4) dynamic routing (3 iterations)
    const int statN = NN * CI * SPATIAL;        // 147,456
    const int pvN   = NN * CO * SPATIAL;        // 294,912
    for (int d = 0; d < 3; ++d) {
        routing_stats_kernel<<<(statN + 255) / 256, 256, 0, stream>>>(b, bmax, cs);
        routing_sumc_kernel <<<(statN + 255) / 256, 256, 0, stream>>>(cs, sumc);
        routing_pv_kernel   <<<(pvN   + 255) / 256, 256, 0, stream>>>(
            uhat, b, bmax, sumc, vout, (d < 2) ? 1 : 0);
    }
}
